// NeuralCDE_5712306504147
// MI455X (gfx1250) — matrix operand order repaired and checked by reading, not compiled, not run on hardware
//
#include <hip/hip_runtime.h>
#include <hip/hip_bf16.h>

// Neural CDE on gfx1250: persistent time-loop kernel.
// 8 batch-groups x 16 workgroups; WG (grp,g) owns h-slice [g*16, g*16+16)
// (= Wf rows [g*512, g*512+512)) for 64 batches. Wf slice lives in LDS as
// pre-swizzled f16 WMMA A-fragments for the whole scan. z state: f32 master
// in owner's LDS, f16 published to a ping-pong global exchange buffer each
// substep, guarded by a release/acquire counter barrier per group.

typedef __attribute__((ext_vector_type(16))) _Float16 v16h;
typedef __attribute__((ext_vector_type(8)))  float    v8f;
typedef _Float16 f16;

#define BB    512
#define TT    256
#define CC    32
#define HH    256
#define OO    16
#define STEPF 0.5f

#define GROUPS 8      // batch groups
#define WGPG   16     // workgroups per group (h-slices)
#define BBLK   64     // batches per group
#define NTHREADS 256  // 8 waves

// ---- LDS layout (bytes), total 325696 <= 327680 (320 KB) ----
#define OFF_WFA    0         // f16 swizzled Wf A-fragments: 32 ntile x 8 kt x 32 lane x 16 f16 = 262144
#define OFF_ZLDS   262144    // f16 z[64][256]   = 32768
#define OFF_ZF32   294912    // f32 z-slice[64][16] = 4096
#define OFF_ZST    299008    // f16 z-slice staging[64][16] = 2048
#define OFF_DX     301056    // f32 dx[64][32]   = 8192
#define OFF_BFDX   309248    // f32 bfdx[64][16] = 4096
#define OFF_BFS    313344    // f32 bf slice[16][32] = 2048
#define OFF_WR     315392    // f16 Wr[16][256]  = 8192
#define OFF_WIS    323584    // f32 Wi slice[16][32] = 2048
#define OFF_BIS    325632    // f32 bi slice[16] = 64
#define LDS_BYTES  325696

__global__ void cde_init(unsigned* cnt) {
    if (threadIdx.x < GROUPS) cnt[threadIdx.x] = 0u;
}

__global__ __launch_bounds__(NTHREADS, 1)
void cde_main(const float* __restrict__ x,  const float* __restrict__ Wf,
              const float* __restrict__ bf, const float* __restrict__ Wi,
              const float* __restrict__ bi, const float* __restrict__ Wr,
              const float* __restrict__ br, float* __restrict__ out,
              f16* __restrict__ zex, unsigned* __restrict__ cnt)
{
    extern __shared__ char lds[];
    const int tid  = threadIdx.x;
    const int lane = tid & 31;
    const int wave = tid >> 5;
    const int grp  = (int)blockIdx.x / WGPG;     // batch group 0..7
    const int g    = (int)blockIdx.x % WGPG;     // h-slice 0..15
    const int b0   = grp * BBLK;                 // global batch base

    f16*   WfA  = (f16*)  (lds + OFF_WFA);
    f16*   zl   = (f16*)  (lds + OFF_ZLDS);
    float* zf   = (float*)(lds + OFF_ZF32);
    f16*   zst  = (f16*)  (lds + OFF_ZST);
    float* dxl  = (float*)(lds + OFF_DX);
    float* bfdx = (float*)(lds + OFF_BFDX);
    float* bfs  = (float*)(lds + OFF_BFS);
    f16*   wr   = (f16*)  (lds + OFF_WR);
    float* wis  = (float*)(lds + OFF_WIS);
    float* bis  = (float*)(lds + OFF_BIS);

    f16*      zbuf = zex + (size_t)grp * (2 * BBLK * HH);  // ping-pong pair
    unsigned* gcnt = cnt + grp;

    // ---------------- setup: stage weights into LDS ----------------
    for (int idx = tid; idx < 16 * CC; idx += NTHREADS) {
        int hl = idx >> 5, c = idx & 31;
        bfs[idx] = bf[(g * 16 + hl) * CC + c];
        wis[idx] = Wi[(g * 16 + hl) * CC + c];
    }
    if (tid < 16) bis[tid] = bi[g * 16 + tid];
    for (int idx = tid; idx < OO * HH; idx += NTHREADS)
        wr[idx] = (f16)Wr[idx];

    // Wf slice -> pre-swizzled A fragments (contiguous k order, same as the B loads):
    // lane l<16: row=l%16, K = {0..15}; lane>=16: K = {16..31}
    for (int ci = tid; ci < 32 * 8 * 32; ci += NTHREADS) {
        int tl   = ci & 31;
        int kt   = (ci >> 5) & 7;
        int nt   = ci >> 8;
        int row  = g * 512 + nt * 16 + (tl & 15);
        int koff = kt * 32 + ((tl < 16) ? 0 : 16);
        const float* src = Wf + (size_t)row * HH + koff;
        f16* dst = WfA + (size_t)ci * 16;
#pragma unroll
        for (int j = 0; j < 8; ++j) {
            dst[j]     = (f16)src[j];
            dst[j + 8] = (f16)src[j + 8];
        }
    }

    // x[:,0,:] into dx buffer (for z0)
    for (int idx = tid; idx < BBLK * CC; idx += NTHREADS) {
        int b = idx >> 5, c = idx & 31;
        dxl[idx] = x[(size_t)(b0 + b) * TT * CC + c];
    }
    __syncthreads();

    // ---------------- z0 = x0 @ Wi.T + bi (slice) ----------------
    for (int idx = tid; idx < BBLK * 16; idx += NTHREADS) {
        int b = idx >> 4, hl = idx & 15;
        float s = bis[hl];
#pragma unroll 8
        for (int c = 0; c < CC; ++c) s += wis[hl * 32 + c] * dxl[b * 32 + c];
        zf[idx]  = s;
        zst[idx] = (f16)s;
    }
    __syncthreads();

    // publish z0 slice into buffer 0
    {
        int b = tid >> 2, part = tid & 3;
        *(uint2*)(zbuf + b * HH + g * 16 + part * 4) =
            *(const uint2*)(zst + b * 16 + part * 4);
    }
    __threadfence();
    __syncthreads();
    unsigned ep = 1;
    if (tid == 0) {
        __hip_atomic_fetch_add(gcnt, 1u, __ATOMIC_RELEASE, __HIP_MEMORY_SCOPE_AGENT);
        while (__hip_atomic_load(gcnt, __ATOMIC_ACQUIRE, __HIP_MEMORY_SCOPE_AGENT) < ep * WGPG)
            __builtin_amdgcn_s_sleep(1);
    }
    __syncthreads();

    int cur = 0;
    // ---------------- time loop: 255 intervals x 2 Euler substeps ----------------
    for (int i = 0; i < TT - 1; ++i) {
        // dx = x[:,i+1,:] - x[:,i,:]
        for (int idx = tid; idx < BBLK * CC; idx += NTHREADS) {
            int b = idx >> 5, c = idx & 31;
            const float* xb = x + (size_t)(b0 + b) * TT * CC + (size_t)i * CC + c;
            dxl[idx] = xb[CC] - xb[0];
        }
        __syncthreads();
        // bfdx[b,hl] = sum_c bf[h*32+c] * dx[b,c]   (same for both substeps)
        for (int idx = tid; idx < BBLK * 16; idx += NTHREADS) {
            int b = idx >> 4, hl = idx & 15;
            float s = 0.f;
#pragma unroll 8
            for (int c = 0; c < CC; ++c) s += bfs[hl * 32 + c] * dxl[b * 32 + c];
            bfdx[idx] = s;
        }

        for (int ss = 0; ss < 2; ++ss) {
            __syncthreads();
            // refill full z (f16) from exchange buffer buf[cur]
            {
                const uint4* src = (const uint4*)(zbuf + (size_t)cur * BBLK * HH);
                uint4*       dst = (uint4*)zl;
#pragma unroll
                for (int r = 0; r < 8; ++r) dst[tid + r * NTHREADS] = src[tid + r * NTHREADS];
            }
            __syncthreads();

            // readout for t=i from the z we just loaded (only once per interval)
            // waves 0..1 whole -> no intra-wave divergence around WMMA
            if (ss == 0 && tid < 64) {
                int bb = g * 4 + (tid >> 4), o = tid & 15;
                float acc = br[o];
#pragma unroll 8
                for (int h = 0; h < HH; ++h)
                    acc += (float)zl[bb * HH + h] * (float)wr[o * HH + h];
                out[(size_t)(b0 + bb) * TT * OO + (size_t)i * OO + o] = acc;
            }

            // ---- GEMM fzT[n,b] = Wf_slice . zT, contract with dx, update z ----
            const int hbase = wave * 2;  // this wave's 2 local h channels
#pragma unroll 1
            for (int bt = 0; bt < 4; ++bt) {
                const int bcol = bt * 16 + (lane & 15);
                v16h Bt[8];
#pragma unroll
                for (int kt = 0; kt < 8; ++kt)
                    Bt[kt] = *(const v16h*)(zl + bcol * HH + kt * 32 + ((lane < 16) ? 0 : 16));
                float dq[2][8];
                {
                    const float* dsrc = dxl + bcol * 32 + ((lane < 16) ? 0 : 8);
#pragma unroll
                    for (int j = 0; j < 8; ++j) { dq[0][j] = dsrc[j]; dq[1][j] = dsrc[16 + j]; }
                }
                float zh[2] = {0.f, 0.f};
#pragma unroll
                for (int mt = 0; mt < 4; ++mt) {
                    v8f acc = {};
#pragma unroll
                    for (int kt = 0; kt < 8; ++kt) {
                        v16h a = *(const v16h*)(WfA + (size_t)((wave * 4 + mt) * 8 + kt) * 512 + lane * 16);
                        acc = __builtin_amdgcn_wmma_f32_16x16x32_f16(
                                  false, a, false, Bt[kt], (short)0, acc, false, false);
                    }
                    float p = 0.f;
#pragma unroll
                    for (int j = 0; j < 8; ++j) p += acc[j] * dq[mt & 1][j];
                    zh[mt >> 1] += p;
                }
                zh[0] += __shfl_xor(zh[0], 16, 32);
                zh[1] += __shfl_xor(zh[1], 16, 32);
                if (lane < 16) {
                    int b = bt * 16 + lane;
#pragma unroll
                    for (int q = 0; q < 2; ++q) {
                        int hl = hbase + q;
                        float zo = zf[b * 16 + hl];
                        float zn = zo + STEPF * (zh[q] + bfdx[b * 16 + hl]);
                        zf[b * 16 + hl]  = zn;
                        zst[b * 16 + hl] = (f16)zn;
                    }
                }
            }
            __syncthreads();
            // publish updated slice into buf[cur^1]
            {
                f16* dst = zbuf + (size_t)(cur ^ 1) * BBLK * HH;
                int b = tid >> 2, part = tid & 3;
                *(uint2*)(dst + b * HH + g * 16 + part * 4) =
                    *(const uint2*)(zst + b * 16 + part * 4);
            }
            __threadfence();
            __syncthreads();
            ++ep;
            if (tid == 0) {
                __hip_atomic_fetch_add(gcnt, 1u, __ATOMIC_RELEASE, __HIP_MEMORY_SCOPE_AGENT);
                while (__hip_atomic_load(gcnt, __ATOMIC_ACQUIRE, __HIP_MEMORY_SCOPE_AGENT) < ep * WGPG)
                    __builtin_amdgcn_s_sleep(1);
            }
            __syncthreads();
            cur ^= 1;
        }
    }

    // ---------------- final readout t = T-1 ----------------
    {
        const uint4* src = (const uint4*)(zbuf + (size_t)cur * BBLK * HH);
        uint4*       dst = (uint4*)zl;
#pragma unroll
        for (int r = 0; r < 8; ++r) dst[tid + r * NTHREADS] = src[tid + r * NTHREADS];
    }
    __syncthreads();
    if (tid < 64) {
        int bb = g * 4 + (tid >> 4), o = tid & 15;
        float acc = br[o];
#pragma unroll 8
        for (int h = 0; h < HH; ++h)
            acc += (float)zl[bb * HH + h] * (float)wr[o * HH + h];
        out[(size_t)(b0 + bb) * TT * OO + (size_t)(TT - 1) * OO + o] = acc;
    }
}

extern "C" void kernel_launch(void* const* d_in, const int* in_sizes, int n_in,
                              void* d_out, int out_size, void* d_ws, size_t ws_size,
                              hipStream_t stream) {
    (void)in_sizes; (void)n_in; (void)out_size; (void)ws_size;
    const float* x  = (const float*)d_in[0];
    const float* Wf = (const float*)d_in[1];
    const float* bf = (const float*)d_in[2];
    const float* Wi = (const float*)d_in[3];
    const float* bi = (const float*)d_in[4];
    const float* Wr = (const float*)d_in[5];
    const float* br = (const float*)d_in[6];
    float* out = (float*)d_out;

    // ws layout: [8 groups][2 ping-pong][64 b][256 h] f16 = 512 KB, then 8 u32 counters
    f16*      zex = (f16*)d_ws;
    unsigned* cnt = (unsigned*)((char*)d_ws + (size_t)GROUPS * 2 * BBLK * HH * sizeof(f16));

    cde_init<<<1, 64, 0, stream>>>(cnt);
    cde_main<<<GROUPS * WGPG, NTHREADS, LDS_BYTES, stream>>>(
        x, Wf, bf, Wi, bi, Wr, br, out, zex, cnt);
}
